// GAT_82867099009055
// MI455X (gfx1250) — compile-verified, matrix operand order checked by reading, not run on hardware
//
#include <hip/hip_runtime.h>
#include <hip/hip_bf16.h>

typedef __attribute__((ext_vector_type(16))) _Float16 v16h;
typedef __attribute__((ext_vector_type(8)))  _Float16 v8h;
typedef __attribute__((ext_vector_type(8)))  float    v8f;

#define GAT_ALPHA 0.2f
#define GAT_BN_EPS 1e-5f

__device__ __forceinline__ float lrelu(float v) { return v > 0.f ? v : GAT_ALPHA * v; }

// monotonic float<->uint encoding for atomicMax over signed floats
__device__ __forceinline__ unsigned fenc(float f) {
    unsigned u = __float_as_uint(f);
    return (u & 0x80000000u) ? ~u : (u | 0x80000000u);
}
__device__ __forceinline__ float fdec(unsigned u) {
    return (u & 0x80000000u) ? __uint_as_float(u & 0x7FFFFFFFu) : __uint_as_float(~u);
}

// ---------------- conversion / transpose kernels ----------------

__global__ void k_cvt_f16(const float* __restrict__ x, _Float16* __restrict__ y, int n) {
    int i = blockIdx.x * blockDim.x + threadIdx.x;
    if (i < n) y[i] = (_Float16)x[i];
}

// W_heads [8][128][64] -> Wt1 [512][128] (col-major B for GEMM), f16
__global__ void k_make_wt1(const float* __restrict__ Wh, _Float16* __restrict__ Wt) {
    int idx = blockIdx.x * blockDim.x + threadIdx.x; // idx = j*128 + k
    if (idx >= 512 * 128) return;
    int j = idx >> 7, k = idx & 127;
    int h = j >> 6, j64 = j & 63;
    Wt[idx] = (_Float16)Wh[h * 8192 + k * 64 + j64];
}

// W [K][Nc] f32 -> Wt [Nc][K] f16
__global__ void k_transpose_cvt(const float* __restrict__ W, _Float16* __restrict__ Wt,
                                int K, int Nc) {
    int idx = blockIdx.x * blockDim.x + threadIdx.x;
    if (idx >= K * Nc) return;
    int k = idx / Nc, j = idx - k * Nc;
    Wt[(size_t)j * K + k] = (_Float16)W[idx];
}

__global__ void k_init_smax(unsigned* __restrict__ smax) {
    if (threadIdx.x < 16) smax[threadIdx.x] = fenc(-3.0e38f);
}

// ---------------- WMMA GEMM: C[M,Ncols] = A[M,K] * B, Bt = B^T [Ncols][K] ----------------
// one wave computes a 16x64 strip (4 N-tiles). All 10 b128 loads of a k-step are
// issued before the 4 WMMAs so the scheduler can clause them and use partial waits.
__global__ void k_gemm_wmma4(const _Float16* __restrict__ A, const _Float16* __restrict__ Bt,
                             float* __restrict__ C, int M, int Ncols, int K, int totalWaves) {
    int gw = blockIdx.x * (blockDim.x >> 5) + (threadIdx.x >> 5);
    if (gw >= totalWaves) return;              // wave-uniform: EXEC all-ones inside
    int lane = threadIdx.x & 31;
    int strips = Ncols >> 6;                   // strips of 64 output columns
    int tm = gw / strips, ts = gw - tm * strips;

    int r  = (tm << 4) + (lane & 15);          // A row for this lane
    int kb = (lane & 16) ? 8 : 0;              // half-wave K base (ISA 16-bit A/B layout)
    const _Float16* ap = A  + (size_t)r * K + kb;
    const _Float16* bp = Bt + (size_t)((ts << 6) + (lane & 15)) * K + kb;
    const size_t bstride = (size_t)16 * K;     // 16 columns of B^T

    v8f acc[4] = {{}, {}, {}, {}};
    for (int kk = 0; kk < K; kk += 32) {
        // ---- issue all loads first ----
        v8h a0 = *(const v8h*)(ap + kk);        // K = kb+0..kb+7
        v8h a1 = *(const v8h*)(ap + kk + 16);   // K = kb+16..kb+23
        v8h b0[4], b1[4];
#pragma unroll
        for (int t = 0; t < 4; ++t) {
            const _Float16* b = bp + (size_t)t * bstride + kk;
            b0[t] = *(const v8h*)(b);
            b1[t] = *(const v8h*)(b + 16);
        }
        // ---- then compute ----
        v16h av;
#pragma unroll
        for (int i = 0; i < 8; ++i) { av[i] = a0[i]; av[i + 8] = a1[i]; }
#pragma unroll
        for (int t = 0; t < 4; ++t) {
            v16h bv;
#pragma unroll
            for (int i = 0; i < 8; ++i) { bv[i] = b0[t][i]; bv[i + 8] = b1[t][i]; }
            acc[t] = __builtin_amdgcn_wmma_f32_16x16x32_f16(
                /*neg_a=*/false, av, /*neg_b=*/false, bv,
                /*c_mod=*/(short)0, acc[t], /*reuse_a=*/false, /*reuse_b=*/false);
        }
    }
    // C/D layout: lane 0-15 -> rows 0..7 in vgpr 0..7 ; lane 16-31 -> rows 8..15
    int row0 = (tm << 4) + ((lane & 16) ? 8 : 0);
    int col0 = (ts << 6) + (lane & 15);
#pragma unroll
    for (int v = 0; v < 8; ++v) {
        float* crow = C + (size_t)(row0 + v) * Ncols + col0;
        crow[0]  = acc[0][v];
        crow[16] = acc[1][v];
        crow[32] = acc[2][v];
        crow[48] = acc[3][v];
    }
}

// ---------------- attention scalar projections ----------------

// msg [N,512]; a [8][128]; as/ad [N*8] (node-major, head minor)
__global__ void k_attn_h8(const float* __restrict__ msg, const float* __restrict__ a,
                          float* __restrict__ as, float* __restrict__ ad, int N) {
    int t = blockIdx.x * blockDim.x + threadIdx.x; // t = i*8 + h
    if (t >= N * 8) return;
    int i = t >> 3, h = t & 7;
    const float* m  = msg + (size_t)i * 512 + h * 64;
    const float* av = a + h * 128;
    float s = 0.f, d = 0.f;
#pragma unroll
    for (int c = 0; c < 64; ++c) { s += m[c] * av[c]; d += m[c] * av[64 + c]; }
    as[t] = s; ad[t] = d;
}

// msg2 [N,128]; a [256]
__global__ void k_attn_1(const float* __restrict__ msg, const float* __restrict__ a,
                         float* __restrict__ as, float* __restrict__ ad, int N) {
    int i = blockIdx.x * blockDim.x + threadIdx.x;
    if (i >= N) return;
    const float* m = msg + (size_t)i * 128;
    float s = 0.f, d = 0.f;
    for (int c = 0; c < 128; ++c) { s += m[c] * a[c]; d += m[c] * a[128 + c]; }
    as[i] = s; ad[i] = d;
}

// ---------------- edge max (segment softmax prep) ----------------

__global__ void k_edgemax_h8(const int* __restrict__ src, const int* __restrict__ dst,
                             const float* __restrict__ as, const float* __restrict__ ad,
                             unsigned* __restrict__ smax, int E) {
    float mx[8];
#pragma unroll
    for (int h = 0; h < 8; ++h) mx[h] = -3.0e38f;
    int stride = gridDim.x * blockDim.x;
    for (int e = blockIdx.x * blockDim.x + threadIdx.x; e < E; e += stride) {
        __builtin_prefetch(src + e + stride, 0, 1);   // -> global_prefetch_b8 (speculative)
        __builtin_prefetch(dst + e + stride, 0, 1);
        int s = src[e], d = dst[e];
#pragma unroll
        for (int h = 0; h < 8; ++h)
            mx[h] = fmaxf(mx[h], lrelu(as[s * 8 + h] + ad[d * 8 + h]));
    }
#pragma unroll
    for (int h = 0; h < 8; ++h)
        for (int off = 16; off; off >>= 1) mx[h] = fmaxf(mx[h], __shfl_xor(mx[h], off, 32));
    if ((threadIdx.x & 31) == 0) {
#pragma unroll
        for (int h = 0; h < 8; ++h) atomicMax(&smax[h], fenc(mx[h]));
    }
}

__global__ void k_edgemax_1(const int* __restrict__ src, const int* __restrict__ dst,
                            const float* __restrict__ as, const float* __restrict__ ad,
                            unsigned* __restrict__ smax, int E) {
    float mx = -3.0e38f;
    int stride = gridDim.x * blockDim.x;
    for (int e = blockIdx.x * blockDim.x + threadIdx.x; e < E; e += stride) {
        __builtin_prefetch(src + e + stride, 0, 1);
        __builtin_prefetch(dst + e + stride, 0, 1);
        mx = fmaxf(mx, lrelu(as[src[e]] + ad[dst[e]]));
    }
    for (int off = 16; off; off >>= 1) mx = fmaxf(mx, __shfl_xor(mx, off, 32));
    if ((threadIdx.x & 31) == 0) atomicMax(smax, fenc(mx));
}

// ---------------- denominator (segment sum of exp) ----------------

__global__ void k_denom_h8(const int* __restrict__ src, const int* __restrict__ dst,
                           const float* __restrict__ as, const float* __restrict__ ad,
                           const unsigned* __restrict__ smax, float* __restrict__ dn, int E) {
    float m[8];
#pragma unroll
    for (int h = 0; h < 8; ++h) m[h] = fdec(smax[h]);
    int stride = gridDim.x * blockDim.x;
    for (int e = blockIdx.x * blockDim.x + threadIdx.x; e < E; e += stride) {
        __builtin_prefetch(src + e + stride, 0, 1);
        __builtin_prefetch(dst + e + stride, 0, 1);
        int s = src[e], d = dst[e];
#pragma unroll
        for (int h = 0; h < 8; ++h) {
            float ev = __expf(lrelu(as[s * 8 + h] + ad[d * 8 + h]) - m[h]);
            atomicAdd(&dn[d * 8 + h], ev);
        }
    }
}

__global__ void k_denom_1(const int* __restrict__ src, const int* __restrict__ dst,
                          const float* __restrict__ as, const float* __restrict__ ad,
                          const unsigned* __restrict__ smax, float* __restrict__ dn, int E) {
    float m = fdec(smax[0]);
    int stride = gridDim.x * blockDim.x;
    for (int e = blockIdx.x * blockDim.x + threadIdx.x; e < E; e += stride) {
        __builtin_prefetch(src + e + stride, 0, 1);
        __builtin_prefetch(dst + e + stride, 0, 1);
        int s = src[e], d = dst[e];
        atomicAdd(&dn[d], __expf(lrelu(as[s] + ad[d]) - m));
    }
}

// ---------------- weighted scatter-aggregate (L2-resident atomics) ----------------

// one wave per (edge, head); 64 channels -> 2 per lane.
// Edge/attention scalars are wave-uniform: force them onto the scalar path.
__global__ void k_agg_h8(const int* __restrict__ src, const int* __restrict__ dst,
                         const float* __restrict__ as, const float* __restrict__ ad,
                         const unsigned* __restrict__ smax, const float* __restrict__ dn,
                         const float* __restrict__ msg, float* __restrict__ out, int E) {
    long gw = ((long)blockIdx.x * blockDim.x + threadIdx.x) >> 5;
    if (gw >= (long)E * 8) return;
    int lane = threadIdx.x & 31;
    int e = __builtin_amdgcn_readfirstlane((int)(gw >> 3));
    int h = __builtin_amdgcn_readfirstlane((int)(gw & 7));
    int s = __builtin_amdgcn_readfirstlane(src[e]);
    int d = __builtin_amdgcn_readfirstlane(dst[e]);
    float v = lrelu(as[s * 8 + h] + ad[d * 8 + h]);
    float coef = __expf(v - fdec(smax[h])) / (dn[d * 8 + h] + 1e-10f);
    const float* m = msg + (size_t)s * 512 + h * 64;
    float*       o = out + (size_t)d * 512 + h * 64;
    atomicAdd(&o[lane],      coef * m[lane]);
    atomicAdd(&o[lane + 32], coef * m[lane + 32]);
}

// one wave per edge; 128 channels -> 4 per lane
__global__ void k_agg_1(const int* __restrict__ src, const int* __restrict__ dst,
                        const float* __restrict__ as, const float* __restrict__ ad,
                        const unsigned* __restrict__ smax, const float* __restrict__ dn,
                        const float* __restrict__ msg, float* __restrict__ out, int E) {
    long gw = ((long)blockIdx.x * blockDim.x + threadIdx.x) >> 5;
    if (gw >= E) return;
    int lane = threadIdx.x & 31;
    int e = __builtin_amdgcn_readfirstlane((int)gw);
    int s = __builtin_amdgcn_readfirstlane(src[e]);
    int d = __builtin_amdgcn_readfirstlane(dst[e]);
    float v = lrelu(as[s] + ad[d]);
    float coef = __expf(v - fdec(smax[0])) / (dn[d] + 1e-10f);
    const float* m = msg + (size_t)s * 128;
    float*       o = out + (size_t)d * 128;
#pragma unroll
    for (int j = 0; j < 4; ++j)
        atomicAdd(&o[lane + 32 * j], coef * m[lane + 32 * j]);
}

// ---------------- BatchNorm (training-mode batch stats) ----------------

// coalesced row-chunk partial sums: LDS float atomics per channel, then one
// global atomicAdd per channel per block. C must be a power of 2, C <= 512.
__global__ void k_bnsum(const float* __restrict__ X, float* __restrict__ sum,
                        float* __restrict__ sumsq, int N, int C, int rowsPerBlock) {
    __shared__ float sh[1024];                  // [0,C) = sum, [C,2C) = sumsq
    for (int i = threadIdx.x; i < 2 * C; i += blockDim.x) sh[i] = 0.f;
    __syncthreads();
    long base = (long)blockIdx.x * rowsPerBlock * C;
    long end  = base + (long)rowsPerBlock * C;
    long lim  = (long)N * C;
    if (end > lim) end = lim;
    for (long idx = base + threadIdx.x; idx < end; idx += blockDim.x) {
        float v = X[idx];
        int c = (int)(idx & (C - 1));
        atomicAdd(&sh[c], v);                   // ds_add_f32, conflict-free lanes
        atomicAdd(&sh[C + c], v * v);
    }
    __syncthreads();
    for (int c = threadIdx.x; c < C; c += blockDim.x) {
        atomicAdd(&sum[c], sh[c]);
        atomicAdd(&sumsq[c], sh[C + c]);
    }
}

__global__ void k_bnfin(const float* __restrict__ sum, const float* __restrict__ sumsq,
                        float* __restrict__ mean, float* __restrict__ rstd, int N, int C) {
    int c = blockIdx.x * blockDim.x + threadIdx.x;
    if (c >= C) return;
    float m = sum[c] / N;
    mean[c] = m;
    rstd[c] = rsqrtf(sumsq[c] / N - m * m + GAT_BN_EPS);
}

// BN apply + f16 convert (feeds layer-2 GEMM); cmask = C-1 (C power of 2)
__global__ void k_bnapply_cvt(const float* __restrict__ X, const float* __restrict__ mean,
                              const float* __restrict__ rstd, const float* __restrict__ gamma,
                              const float* __restrict__ beta, _Float16* __restrict__ Y,
                              int total, int cmask) {
    int i = blockIdx.x * blockDim.x + threadIdx.x;
    if (i >= total) return;
    int c = i & cmask;
    Y[i] = (_Float16)((X[i] - mean[c]) * rstd[c] * gamma[c] + beta[c]);
}

// final: BN2 + residual (feat@W_res, precomputed) + b_res
__global__ void k_final(const float* __restrict__ X, const float* __restrict__ mean,
                        const float* __restrict__ rstd, const float* __restrict__ gamma,
                        const float* __restrict__ beta, const float* __restrict__ res,
                        const float* __restrict__ br, float* __restrict__ Y, int total) {
    int i = blockIdx.x * blockDim.x + threadIdx.x;
    if (i >= total) return;
    int c = i & 127;
    Y[i] = (X[i] - mean[c]) * rstd[c] * gamma[c] + beta[c] + res[i] + br[c];
}

// ---------------- launch ----------------

extern "C" void kernel_launch(void* const* d_in, const int* in_sizes, int n_in,
                              void* d_out, int out_size, void* d_ws, size_t ws_size,
                              hipStream_t stream) {
    (void)in_sizes; (void)n_in; (void)out_size; (void)ws_size;
    const int N = 50000, E = 1600000;

    const float* feat  = (const float*)d_in[0];
    const int*   edges = (const int*)d_in[1];
    const float* Wh    = (const float*)d_in[2];
    const float* ah    = (const float*)d_in[3];
    const float* gam_h = (const float*)d_in[4];   // [8,64] flat == 512 channels
    const float* bet_h = (const float*)d_in[5];
    const float* Wo    = (const float*)d_in[6];
    const float* ao    = (const float*)d_in[7];
    const float* gam_o = (const float*)d_in[8];
    const float* bet_o = (const float*)d_in[9];
    const float* Wr    = (const float*)d_in[10];
    const float* br    = (const float*)d_in[11];
    const int* src = edges;
    const int* dst = edges + E;

    char* ws = (char*)d_ws;
    size_t off = 0;
    auto carve = [&](size_t bytes) -> void* {
        void* p = ws + off;
        off = (off + bytes + 255) & ~(size_t)255;
        return p;
    };

    _Float16* feat16 = (_Float16*)carve((size_t)N * 128 * 2);
    _Float16* wt1    = (_Float16*)carve(512 * 128 * 2);
    _Float16* wt2    = (_Float16*)carve(128 * 512 * 2);
    _Float16* wrt    = (_Float16*)carve(128 * 128 * 2);
    float*    msg1   = (float*)carve((size_t)N * 512 * 4);
    float*    out1   = (float*)carve((size_t)N * 512 * 4);
    _Float16* x16    = (_Float16*)carve((size_t)N * 512 * 2);
    float*    msg2   = (float*)carve((size_t)N * 128 * 4);
    float*    out2   = (float*)carve((size_t)N * 128 * 4);
    float*    res    = (float*)carve((size_t)N * 128 * 4);
    float*    as1    = (float*)carve((size_t)N * 8 * 4);
    float*    ad1    = (float*)carve((size_t)N * 8 * 4);
    float*    dn1    = (float*)carve((size_t)N * 8 * 4);
    float*    as2    = (float*)carve((size_t)N * 4);
    float*    ad2    = (float*)carve((size_t)N * 4);
    float*    dn2    = (float*)carve((size_t)N * 4);
    float*    sum1   = (float*)carve(512 * 4);
    float*    sq1    = (float*)carve(512 * 4);
    float*    sum2   = (float*)carve(128 * 4);
    float*    sq2    = (float*)carve(128 * 4);
    float*    mean1  = (float*)carve(512 * 4);
    float*    rstd1  = (float*)carve(512 * 4);
    float*    mean2  = (float*)carve(128 * 4);
    float*    rstd2  = (float*)carve(128 * 4);
    unsigned* smax   = (unsigned*)carve(16 * 4);

    // operand prep
    k_cvt_f16<<<(N * 128 + 255) / 256, 256, 0, stream>>>(feat, feat16, N * 128);
    k_make_wt1<<<(512 * 128 + 255) / 256, 256, 0, stream>>>(Wh, wt1);
    k_transpose_cvt<<<(512 * 128 + 255) / 256, 256, 0, stream>>>(Wo, wt2, 512, 128);
    k_transpose_cvt<<<(128 * 128 + 255) / 256, 256, 0, stream>>>(Wr, wrt, 128, 128);
    k_init_smax<<<1, 32, 0, stream>>>(smax);

    const int rpb = 128;                            // rows per bnsum block
    const int bnBlocks = (N + rpb - 1) / rpb;

    // ---- layer 1 (8 heads fused: msg1 = feat @ [W_h0|...|W_h7]) ----
    {
        int tw = (N / 16) * (512 / 64);             // 16x64 strips
        k_gemm_wmma4<<<(tw + 7) / 8, 256, 0, stream>>>(feat16, wt1, msg1, N, 512, 128, tw);
    }
    k_attn_h8<<<(N * 8 + 255) / 256, 256, 0, stream>>>(msg1, ah, as1, ad1, N);
    k_edgemax_h8<<<2048, 256, 0, stream>>>(src, dst, as1, ad1, smax, E);
    (void)hipMemsetAsync(dn1, 0, (size_t)N * 8 * 4, stream);
    k_denom_h8<<<2048, 256, 0, stream>>>(src, dst, as1, ad1, smax, dn1, E);
    (void)hipMemsetAsync(out1, 0, (size_t)N * 512 * 4, stream);
    {
        long waves = (long)E * 8;
        int blocks = (int)((waves + 7) / 8);
        k_agg_h8<<<blocks, 256, 0, stream>>>(src, dst, as1, ad1, smax, dn1, msg1, out1, E);
    }
    (void)hipMemsetAsync(sum1, 0, 512 * 4, stream);
    (void)hipMemsetAsync(sq1, 0, 512 * 4, stream);
    k_bnsum<<<bnBlocks, 256, 0, stream>>>(out1, sum1, sq1, N, 512, rpb);
    k_bnfin<<<2, 256, 0, stream>>>(sum1, sq1, mean1, rstd1, N, 512);
    k_bnapply_cvt<<<(N * 512 + 255) / 256, 256, 0, stream>>>(out1, mean1, rstd1, gam_h, bet_h,
                                                             x16, N * 512, 511);

    // ---- layer 2 (msg2 = x @ W_out) ----
    {
        int tw = (N / 16) * (128 / 64);
        k_gemm_wmma4<<<(tw + 7) / 8, 256, 0, stream>>>(x16, wt2, msg2, N, 128, 512, tw);
    }
    k_attn_1<<<(N + 255) / 256, 256, 0, stream>>>(msg2, ao, as2, ad2, N);
    k_edgemax_1<<<2048, 256, 0, stream>>>(src, dst, as2, ad2, smax + 8, E);
    (void)hipMemsetAsync(dn2, 0, (size_t)N * 4, stream);
    k_denom_1<<<2048, 256, 0, stream>>>(src, dst, as2, ad2, smax + 8, dn2, E);
    (void)hipMemsetAsync(out2, 0, (size_t)N * 128 * 4, stream);
    {
        int blocks = E / 8; // one wave per edge, 8 waves per block
        k_agg_1<<<blocks, 256, 0, stream>>>(src, dst, as2, ad2, smax + 8, dn2, msg2, out2, E);
    }
    (void)hipMemsetAsync(sum2, 0, 128 * 4, stream);
    (void)hipMemsetAsync(sq2, 0, 128 * 4, stream);
    k_bnsum<<<bnBlocks, 256, 0, stream>>>(out2, sum2, sq2, N, 128, rpb);
    k_bnfin<<<1, 128, 0, stream>>>(sum2, sq2, mean2, rstd2, N, 128);

    // ---- residual GEMM + final fuse ----
    {
        int tw = (N / 16) * (128 / 64);
        k_gemm_wmma4<<<(tw + 7) / 8, 256, 0, stream>>>(feat16, wrt, res, N, 128, 128, tw);
    }
    k_final<<<(N * 128 + 255) / 256, 256, 0, stream>>>(out2, mean2, rstd2, gam_o, bet_o,
                                                       res, br, (float*)d_out, N * 128);
}
